// ReDownSample_14886356648205
// MI455X (gfx1250) — matrix operand order repaired and checked by reading, not compiled, not run on hardware
//
#include <hip/hip_runtime.h>
#include <hip/hip_bf16.h>

typedef __attribute__((ext_vector_type(8)))  float        v8f;
typedef __attribute__((ext_vector_type(16))) __bf16       v16bf;
typedef __attribute__((ext_vector_type(2)))  unsigned int u32x2;
typedef __attribute__((ext_vector_type(4)))  unsigned int u32x4;

union BF16FragU { u32x4 q[2]; u32x2 d[4]; unsigned int w[8]; v16bf v; };

#define C_    128
#define Hc_   256
#define Wc_   256
#define NPIX  65536   // B * 128 * 128
#define NG_   8

__device__ inline unsigned short f2bf(float f) {
  unsigned u = __float_as_uint(f);
  unsigned r = u + 0x7FFFu + ((u >> 16) & 1u);   // round-to-nearest-even
  return (unsigned short)(r >> 16);
}
__device__ inline unsigned int pack2(float a, float b) {
  return (unsigned int)f2bf(a) | ((unsigned int)f2bf(b) << 16);
}
__device__ inline float bf2f(unsigned short h) {
  return __uint_as_float(((unsigned)h) << 16);
}

// A-frag (16x32 bf16) from pre-converted bf16 weights (pair-packed uints).
// Code's own k order (same as B-frag): lane m=lane&15; uint idx v: hi*8+v (contiguous k = 16*hi + s)
// -> two contiguous 16B chunks per lane.
__device__ inline v16bf load_a_frag_bf(const unsigned int* __restrict__ w32,
                                       int rowstride, int row0, int k0half, int lane) {
  const int m = lane & 15, hi = lane >> 4;
  const unsigned int* p = w32 + (size_t)(row0 + m) * rowstride + k0half + hi * 8;
  BF16FragU f;
  f.q[0] = *(const u32x4*)(p);
  f.q[1] = *(const u32x4*)(p + 4);
  return f.v;
}

// B-frag (32x16 bf16) from LDS laid out [n][kpair] (pair-packed uints):
// lane n = n0+(lane&15); kpair base = k0half + hi*8, 8 contiguous uints.
__device__ inline v16bf load_b_frag_lds(const unsigned int* __restrict__ lds,
                                        int pitch, int k0half, int n0, int lane) {
  const int n = n0 + (lane & 15), hi = lane >> 4;
  const unsigned int* p = lds + n * pitch + k0half + hi * 8;
  BF16FragU f;
#pragma unroll
  for (int j = 0; j < 4; ++j) f.d[j] = *(const u32x2*)(p + 2 * j);
  return f.v;
}

// ---------------- Kernel 0: weight bf16 pre-conversion + BN folding ----------------------
__global__ __launch_bounds__(256)
void k_prep(const float* __restrict__ hfw, const float* __restrict__ lfw,
            const float* __restrict__ hdw, const float* __restrict__ ldw,
            const float* __restrict__ hsw, const float* __restrict__ lsw,
            const float* __restrict__ hfb, const float* __restrict__ hbg,
            const float* __restrict__ hbb, const float* __restrict__ hbm,
            const float* __restrict__ hbv,
            const float* __restrict__ lfb2, const float* __restrict__ lbg,
            const float* __restrict__ lbb, const float* __restrict__ lbm,
            const float* __restrict__ lbv,
            unsigned short* __restrict__ hfw_bf, unsigned short* __restrict__ lfw_bf,
            unsigned short* __restrict__ wc_bf,
            float* __restrict__ bnA_HF, float* __restrict__ bnB_HF,
            float* __restrict__ bnA_LF, float* __restrict__ bnB_LF)
{
  if (blockIdx.x < 384) {
    int j = blockIdx.x * 256 + threadIdx.x;
    if (j < 49152) {
      hfw_bf[j] = f2bf(hfw[j]);
    } else if (j < 81920) {
      int j1 = j - 49152;
      lfw_bf[j1] = f2bf(lfw[j1]);
    } else {
      int j2 = j - 81920;            // 64 x 256 combined [dir;gate] x [hf|lf]
      int m = j2 >> 8, k = j2 & 255;
      float v;
      if (m < 32) v = (k < 128) ? hdw[m * 128 + k] : ldw[m * 128 + (k - 128)];
      else        v = (k < 128) ? hsw[(m - 32) * 128 + k] : lsw[(m - 32) * 128 + (k - 128)];
      wc_bf[j2] = f2bf(v);
    }
  } else {
    int t = threadIdx.x;
    if (t < 128) {
      float sc = hbg[t] * rsqrtf(hbv[t] + 1e-5f);
      bnA_HF[t] = sc;
      bnB_HF[t] = hfb[t] * sc + hbb[t] - hbm[t] * sc;
    } else if (t < 256) {
      int c = t - 128;
      float sc = lbg[c] * rsqrtf(lbv[c] + 1e-5f);
      bnA_LF[c] = sc;
      bnB_LF[c] = lfb2[c] * sc + lbb[c] - lbm[c] * sc;
    }
  }
}

// ---------------- Kernel 1: maxpool + Haar DWT + fused GEMMs + BN/ReLU + GN partials -----
__global__ __launch_bounds__(256)
void k_dwt_fuse(const float* __restrict__ x,
                const unsigned short* __restrict__ hfw_bf,
                const unsigned short* __restrict__ lfw_bf,
                const float* __restrict__ bnA_HF, const float* __restrict__ bnB_HF,
                const float* __restrict__ bnA_LF, const float* __restrict__ bnB_LF,
                unsigned short* __restrict__ hf_ws, unsigned short* __restrict__ lf_ws,
                float* __restrict__ statsHF, float* __restrict__ statsLF)
{
  extern __shared__ unsigned int smem[];
  unsigned int* hfB = smem;               // [64 n][194] pair-packed, kpairs 0..191
  unsigned int* lfB = smem + 64 * 194;    // [64 n][130] pair-packed, kpairs 0..127

  const int tid  = threadIdx.x;
  const int lane = tid & 31;
  const int wave = tid >> 5;
  const int pix0 = blockIdx.x * 64;
  const int b    = pix0 >> 14;
  const int rem  = pix0 & 16383;
  const int oh   = rem >> 7;
  const int ow0  = rem & 127;

  // Phase 1: DWT + maxpool for channel pairs, pack bf16 pairs into LDS dwords.
  for (int i = tid; i < 64 * 64; i += 256) {
    int cp = i >> 6, p = i & 63;         // channel-pair, pixel
    const float* px0 = x + (((size_t)(b * C_ + 2 * cp) * Hc_ + 2 * oh) * Wc_ + 2 * (ow0 + p));
    const float* px1 = px0 + (size_t)Hc_ * Wc_;
    float2 t0 = *(const float2*)px0, b0 = *(const float2*)(px0 + Wc_);
    float2 t1 = *(const float2*)px1, b1 = *(const float2*)(px1 + Wc_);
    float ll0 = (t0.x + t0.y + b0.x + b0.y) * 0.5f;
    float lh0 = (t0.x + t0.y - b0.x - b0.y) * 0.5f;
    float hl0 = (t0.x - t0.y + b0.x - b0.y) * 0.5f;
    float hh0 = (t0.x - t0.y - b0.x + b0.y) * 0.5f;
    float mx0 = fmaxf(fmaxf(t0.x, t0.y), fmaxf(b0.x, b0.y));
    float ll1 = (t1.x + t1.y + b1.x + b1.y) * 0.5f;
    float lh1 = (t1.x + t1.y - b1.x - b1.y) * 0.5f;
    float hl1 = (t1.x - t1.y + b1.x - b1.y) * 0.5f;
    float hh1 = (t1.x - t1.y - b1.x + b1.y) * 0.5f;
    float mx1 = fmaxf(fmaxf(t1.x, t1.y), fmaxf(b1.x, b1.y));
    unsigned int* hrow = hfB + p * 194;
    unsigned int* lrow = lfB + p * 130;
    hrow[cp]       = pack2(lh0, lh1);    // K = c        (LH)
    hrow[64 + cp]  = pack2(hl0, hl1);    // K = 128 + c  (HL)
    hrow[128 + cp] = pack2(hh0, hh1);    // K = 256 + c  (HH)
    lrow[cp]       = pack2(ll0, ll1);    // K = c        (LL)
    lrow[64 + cp]  = pack2(mx0, mx1);    // K = 128 + c  (maxpool)
  }
  __syncthreads();

  const v8f vzero = {0.f, 0.f, 0.f, 0.f, 0.f, 0.f, 0.f, 0.f};
  const int hi = lane >> 4, nn = lane & 15;
  const unsigned int* wHF = (const unsigned int*)hfw_bf;   // row stride 192 uints
  const unsigned int* wLF = (const unsigned int*)lfw_bf;   // row stride 128 uints

  // Phase 2a: HF GEMM (K = 384).
  {
    v8f acc[4];
#pragma unroll
    for (int nt = 0; nt < 4; ++nt) acc[nt] = vzero;
    for (int ks = 0; ks < 12; ++ks) {
      v16bf af = load_a_frag_bf(wHF, 192, wave * 16, ks * 16, lane);
#pragma unroll
      for (int nt = 0; nt < 4; ++nt) {
        v16bf bf = load_b_frag_lds(hfB, 194, ks * 16, nt * 16, lane);
        acc[nt] = __builtin_amdgcn_wmma_f32_16x16x32_bf16(
            false, af, false, bf, (short)0, acc[nt], false, false);
      }
    }
    float s0 = 0.f, s1 = 0.f;
#pragma unroll
    for (int r = 0; r < 8; ++r) {
      int ch = wave * 16 + r + hi * 8;
      float a = bnA_HF[ch], bb = bnB_HF[ch];
#pragma unroll
      for (int nt = 0; nt < 4; ++nt) {
        float v = fmaxf(acc[nt][r] * a + bb, 0.f);
        s0 += v; s1 += v * v;
        hf_ws[(size_t)ch * NPIX + pix0 + nt * 16 + nn] = f2bf(v);
      }
    }
    for (int off = 16; off; off >>= 1) { s0 += __shfl_xor(s0, off); s1 += __shfl_xor(s1, off); }
    if (lane == 0) {
      atomicAdd(&statsHF[(b * NG_ + wave) * 2 + 0], s0);
      atomicAdd(&statsHF[(b * NG_ + wave) * 2 + 1], s1);
    }
  }

  // Phase 2b: LF GEMM (K = 256).
  {
    v8f acc[4];
#pragma unroll
    for (int nt = 0; nt < 4; ++nt) acc[nt] = vzero;
    for (int ks = 0; ks < 8; ++ks) {
      v16bf af = load_a_frag_bf(wLF, 128, wave * 16, ks * 16, lane);
#pragma unroll
      for (int nt = 0; nt < 4; ++nt) {
        v16bf bf = load_b_frag_lds(lfB, 130, ks * 16, nt * 16, lane);
        acc[nt] = __builtin_amdgcn_wmma_f32_16x16x32_bf16(
            false, af, false, bf, (short)0, acc[nt], false, false);
      }
    }
    float s0 = 0.f, s1 = 0.f;
#pragma unroll
    for (int r = 0; r < 8; ++r) {
      int ch = wave * 16 + r + hi * 8;
      float a = bnA_LF[ch], bb = bnB_LF[ch];
#pragma unroll
      for (int nt = 0; nt < 4; ++nt) {
        float v = fmaxf(acc[nt][r] * a + bb, 0.f);
        s0 += v; s1 += v * v;
        lf_ws[(size_t)ch * NPIX + pix0 + nt * 16 + nn] = f2bf(v);
      }
    }
    for (int off = 16; off; off >>= 1) { s0 += __shfl_xor(s0, off); s1 += __shfl_xor(s1, off); }
    if (lane == 0) {
      atomicAdd(&statsLF[(b * NG_ + wave) * 2 + 0], s0);
      atomicAdd(&statsLF[(b * NG_ + wave) * 2 + 1], s1);
    }
  }
}

// ---------------- Kernel 2: finalize GN stats, fold into per-(b,ch) fma coefficients -----
__global__ __launch_bounds__(1024)
void k_gn_stats(const float* __restrict__ statsHF, const float* __restrict__ statsLF,
                const float* __restrict__ ghg, const float* __restrict__ ghb,
                const float* __restrict__ glg, const float* __restrict__ glb,
                float* __restrict__ gnA_HF, float* __restrict__ gnB_HF,
                float* __restrict__ gnA_LF, float* __restrict__ gnB_LF)
{
  __shared__ float mu[64], rs[64];   // 0..31 hf groups, 32..63 lf groups
  int t = threadIdx.x;
  const float invN = 1.f / (16.f * 16384.f);
  if (t < 64) {
    const float* st = (t < 32) ? statsHF : statsLF;
    int u = t & 31;
    float s = st[u * 2], q = st[u * 2 + 1];
    float m = s * invN, v = q * invN - m * m;
    mu[t] = m; rs[t] = rsqrtf(v + 1e-5f);
  }
  __syncthreads();
  if (t < 512) {
    int b = t >> 7, ch = t & 127, g = b * NG_ + (ch >> 4);
    float a = rs[g] * ghg[ch];
    gnA_HF[t] = a; gnB_HF[t] = ghb[ch] - mu[g] * a;
  } else {
    int u = t - 512, b = u >> 7, ch = u & 127, g = 32 + b * NG_ + (ch >> 4);
    float a = rs[g] * glg[ch];
    gnA_LF[u] = a; gnB_LF[u] = glb[ch] - mu[g] * a;
  }
}

// ---------------- Kernel 3: GN-on-load + combined dir/gate GEMM + sigmoid + offsets ------
__global__ __launch_bounds__(256)
void k_dir_gate(const unsigned short* __restrict__ hf_ws, const unsigned short* __restrict__ lf_ws,
                const unsigned short* __restrict__ wc_bf,
                const float* __restrict__ gnA_HF, const float* __restrict__ gnB_HF,
                const float* __restrict__ gnA_LF, const float* __restrict__ gnB_LF,
                const float* __restrict__ hdb, const float* __restrict__ ldb,
                const float* __restrict__ hsb, const float* __restrict__ lsb,
                float* __restrict__ off_ws)
{
  __shared__ unsigned int Bm[32 * 130];   // [n][kpair], K = 256
  __shared__ float combo[64 * 32];        // rows 0..31 dir, 32..63 gate-logit
  const int tid  = threadIdx.x;
  const int lane = tid & 31;
  const int wave = tid >> 5;
  const int pix0 = blockIdx.x * 32;
  const int b    = pix0 >> 14;

  for (int i = tid; i < 128 * 32; i += 256) {
    int kp = i >> 5, n = i & 31;
    int pixG = pix0 + n;
    int k = 2 * kp;
    float v0, v1;
    if (k < 128) {
      int base = b * 128 + k;
      v0 = bf2f(hf_ws[(size_t)k * NPIX + pixG])       * gnA_HF[base]     + gnB_HF[base];
      v1 = bf2f(hf_ws[(size_t)(k + 1) * NPIX + pixG]) * gnA_HF[base + 1] + gnB_HF[base + 1];
    } else {
      int ch = k - 128;
      int base = b * 128 + ch;
      v0 = bf2f(lf_ws[(size_t)ch * NPIX + pixG])        * gnA_LF[base]     + gnB_LF[base];
      v1 = bf2f(lf_ws[(size_t)(ch + 1) * NPIX + pixG])  * gnA_LF[base + 1] + gnB_LF[base + 1];
    }
    Bm[n * 130 + kp] = pack2(v0, v1);
  }
  __syncthreads();

  {
    const int mtile = wave & 3;
    const int nt    = wave >> 2;
    const int mrow0 = mtile * 16;
    const v8f vzero = {0.f, 0.f, 0.f, 0.f, 0.f, 0.f, 0.f, 0.f};
    v8f acc = vzero;
    const unsigned int* wc32 = (const unsigned int*)wc_bf;  // row stride 128 uints
    for (int ks = 0; ks < 8; ++ks) {
      v16bf af = load_a_frag_bf(wc32, 128, mrow0, ks * 16, lane);
      v16bf bf = load_b_frag_lds(Bm, 130, ks * 16, nt * 16, lane);
      acc = __builtin_amdgcn_wmma_f32_16x16x32_bf16(
          false, af, false, bf, (short)0, acc, false, false);
    }
    const int hi = lane >> 4, nn = lane & 15;
#pragma unroll
    for (int r = 0; r < 8; ++r) {
      int mrow = mrow0 + r + hi * 8;
      float bias = (mrow < 32) ? (hdb[mrow] + ldb[mrow]) : (hsb[mrow - 32] + lsb[mrow - 32]);
      combo[mrow * 32 + nt * 16 + nn] = acc[r] + bias;
    }
  }
  __syncthreads();

  for (int i = tid; i < 32 * 32; i += 256) {
    int oc = i >> 5, n = i & 31;
    float d = combo[oc * 32 + n];
    float s = combo[(oc + 32) * 32 + n];
    float gate = 1.f / (1.f + expf(-s));
    int j = oc & 3;
    float ip = (oc < 16) ? ((j & 1) ? 0.25f : -0.25f)
                         : ((j & 2) ? 0.25f : -0.25f);
    off_ws[(size_t)(b * 32 + oc) * 16384 + ((pix0 + n) & 16383)] = d * gate + ip;
  }
}

// ---------------- Kernel 4: deformable bilinear gather (border clamp) --------------------
__global__ __launch_bounds__(256)
void k_sample(const float* __restrict__ x, const float* __restrict__ off_ws,
              float* __restrict__ out)
{
  const int w  = threadIdx.x;          // 0..255
  const int id = blockIdx.x;           // b*1024 + g*256 + h
  const int h  = id & 255;
  const int g  = (id >> 8) & 3;
  const int b  = id >> 10;
  const int oh = h >> 1, ow = w >> 1;
  const int k  = g * 4 + ((h & 1) << 1) + (w & 1);
  const int pidx = oh * 128 + ow;
  float offx = off_ws[(size_t)(b * 32 + k) * 16384 + pidx];
  float offy = off_ws[(size_t)(b * 32 + 16 + k) * 16384 + pidx];
  float ix = fminf(fmaxf(2.f * ow + offx, 0.f), 255.f);
  float iy = fminf(fmaxf(2.f * oh + offy, 0.f), 255.f);
  float x0f = floorf(ix), y0f = floorf(iy);
  int x0 = (int)x0f, y0 = (int)y0f;
  int x1 = (x0 + 1 < 255) ? x0 + 1 : 255;
  int y1 = (y0 + 1 < 255) ? y0 + 1 : 255;
  float wx = ix - x0f, wy = iy - y0f;
  float w00 = (1.f - wx) * (1.f - wy), w01 = wx * (1.f - wy);
  float w10 = (1.f - wx) * wy,         w11 = wx * wy;
  const int a00 = y0 * 256 + x0, a01 = y0 * 256 + x1;
  const int a10 = y1 * 256 + x0, a11 = y1 * 256 + x1;
  const float* xb = x + ((size_t)(b * 128 + g * 32) << 16);
  float* ob = out + (((size_t)(b * 4 + g) * 32) << 16) + h * 256 + w;
#pragma unroll 4
  for (int c = 0; c < 32; ++c) {
    const float* xc = xb + ((size_t)c << 16);
    float v = xc[a00] * w00 + xc[a01] * w01 + xc[a10] * w10 + xc[a11] * w11;
    ob[(size_t)c << 16] = v;
  }
}

extern "C" void kernel_launch(void* const* d_in, const int* in_sizes, int n_in,
                              void* d_out, int out_size, void* d_ws, size_t ws_size,
                              hipStream_t stream)
{
  const float* x   = (const float*)d_in[0];
  const float* hfw = (const float*)d_in[1];
  const float* hfb = (const float*)d_in[2];
  const float* hbg = (const float*)d_in[3];
  const float* hbb = (const float*)d_in[4];
  const float* hbm = (const float*)d_in[5];
  const float* hbv = (const float*)d_in[6];
  const float* lfw = (const float*)d_in[7];
  const float* lfb = (const float*)d_in[8];
  const float* lbg = (const float*)d_in[9];
  const float* lbb = (const float*)d_in[10];
  const float* lbm = (const float*)d_in[11];
  const float* lbv = (const float*)d_in[12];
  const float* ghg = (const float*)d_in[13];
  const float* ghb = (const float*)d_in[14];
  const float* glg = (const float*)d_in[15];
  const float* glb = (const float*)d_in[16];
  const float* hdw = (const float*)d_in[17];
  const float* hdb = (const float*)d_in[18];
  const float* ldw = (const float*)d_in[19];
  const float* ldb = (const float*)d_in[20];
  const float* hsw = (const float*)d_in[21];
  const float* hsb = (const float*)d_in[22];
  const float* lsw = (const float*)d_in[23];
  const float* lsb = (const float*)d_in[24];

  char* ws = (char*)d_ws;
  float* statsHF = (float*)(ws + 0);
  float* statsLF = (float*)(ws + 256);
  float* off_ws  = (float*)(ws + 1024);
  unsigned short* hf_ws = (unsigned short*)(ws + 1024 + 8ull * 1024 * 1024);
  unsigned short* lf_ws = (unsigned short*)(ws + 1024 + 24ull * 1024 * 1024);
  char* wregion = ws + 1024 + 40ull * 1024 * 1024;        // 16B aligned
  unsigned short* hfw_bf = (unsigned short*)(wregion);            // 98304 B
  unsigned short* lfw_bf = (unsigned short*)(wregion + 98304);    // 65536 B
  unsigned short* wc_bf  = (unsigned short*)(wregion + 163840);   // 32768 B
  float* bnA_HF = (float*)(wregion + 196608);
  float* bnB_HF = (float*)(wregion + 197120);
  float* bnA_LF = (float*)(wregion + 197632);
  float* bnB_LF = (float*)(wregion + 198144);
  float* gnA_HF = (float*)(wregion + 198656);
  float* gnB_HF = (float*)(wregion + 200704);
  float* gnA_LF = (float*)(wregion + 202752);
  float* gnB_LF = (float*)(wregion + 204800);

  hipMemsetAsync(ws, 0, 1024, stream);

  k_prep<<<dim3(385), dim3(256), 0, stream>>>(
      hfw, lfw, hdw, ldw, hsw, lsw,
      hfb, hbg, hbb, hbm, hbv, lfb, lbg, lbb, lbm, lbv,
      hfw_bf, lfw_bf, wc_bf, bnA_HF, bnB_HF, bnA_LF, bnB_LF);

  k_dwt_fuse<<<dim3(1024), dim3(256), (64 * 194 + 64 * 130) * sizeof(unsigned int), stream>>>(
      x, hfw_bf, lfw_bf, bnA_HF, bnB_HF, bnA_LF, bnB_LF,
      hf_ws, lf_ws, statsHF, statsLF);

  k_gn_stats<<<dim3(1), dim3(1024), 0, stream>>>(
      statsHF, statsLF, ghg, ghb, glg, glb, gnA_HF, gnB_HF, gnA_LF, gnB_LF);

  k_dir_gate<<<dim3(2048), dim3(256), 0, stream>>>(
      hf_ws, lf_ws, wc_bf, gnA_HF, gnB_HF, gnA_LF, gnB_LF,
      hdb, ldb, hsb, lsb, off_ws);

  k_sample<<<dim3(4096), dim3(256), 0, stream>>>(x, off_ws, (float*)d_out);
}
